// PAM_Module_1726576855892
// MI455X (gfx1250) — matrix operand order repaired and checked by reading, not compiled, not run on hardware
//
#include <hip/hip_runtime.h>
#include <hip/hip_bf16.h>
#include <math.h>

#define BATCH 4
#define CH    512
#define C8v   64
#define NN    4096   // H*W = 64*64

typedef __bf16 bf16;
typedef bf16  bf16x8  __attribute__((ext_vector_type(8)));
typedef bf16  bf16x16 __attribute__((ext_vector_type(16)));
typedef float f32x8   __attribute__((ext_vector_type(8)));

static __device__ __forceinline__ f32x8 wmma_bf16(bf16x16 a, bf16x16 b, f32x8 c) {
  // v_wmma_f32_16x16x32_bf16  (neg_a, A, neg_b, B, c_mod, C, reuse_a, reuse_b)
  return __builtin_amdgcn_wmma_f32_16x16x32_bf16(false, a, false, b, (short)0, c, false, false);
}

static __device__ __forceinline__ f32x8 zero8() {
  f32x8 z = {0.f,0.f,0.f,0.f,0.f,0.f,0.f,0.f};
  return z;
}

// A fragment (16x32 bf16). Lane = hi*16 + r holds row r:
//   halves 0..7  <- row[k0 + hi*16 + 0..7]
//   halves 8..15 <- row[k0 + hi*16 + 8 + 0..7]
static __device__ __forceinline__ bf16x16 load_afrag(const bf16* rowp, int hi) {
  bf16x8 lo = *(const bf16x8*)(rowp + hi*16);
  bf16x8 hv = *(const bf16x8*)(rowp + hi*16 + 8);
  bf16x16 r;
#pragma unroll
  for (int j = 0; j < 8; ++j) { r[j] = lo[j]; r[8+j] = hv[j]; }
  return r;
}

// Same interleave, source is f32 (weights), converted to bf16 on the fly.
static __device__ __forceinline__ bf16x16 load_afrag_f32(const float* rowp, int hi) {
  const float* p0 = rowp + hi*16;
  const float* p1 = rowp + hi*16 + 8;
  bf16x16 r;
#pragma unroll
  for (int j = 0; j < 8; ++j) { r[j] = (bf16)p0[j]; r[8+j] = (bf16)p1[j]; }
  return r;
}

// B fragment (32x16 bf16). Lane = hi*16 + n holds column n, k = k0 + hi*16 + 0..15,
// contiguous 16 bf16 per lane when the source is stored column-major-in-k.
static __device__ __forceinline__ bf16x16 load_bfrag(const bf16* p) {
  bf16x8 a = *(const bf16x8*)(p);
  bf16x8 b = *(const bf16x8*)(p + 8);
  bf16x16 r;
#pragma unroll
  for (int j = 0; j < 8; ++j) { r[j] = a[j]; r[8+j] = b[j]; }
  return r;
}

// ---------------------------------------------------------------------------
// Kernel 1: QKV 1x1-conv projections (GEMM over channels), f32 -> bf16 outputs
//   Qb[b][n][o] , Kb[b][n][o]  (o contiguous -> fragment-friendly for energy)
//   Vb[b][o][n]                (n contiguous -> fragment-friendly for out GEMM)
// ---------------------------------------------------------------------------
__global__ void __launch_bounds__(256)
qkv_proj_kernel(const float* __restrict__ x,
                const float* __restrict__ wq, const float* __restrict__ bq,
                const float* __restrict__ wk, const float* __restrict__ bk,
                const float* __restrict__ wv, const float* __restrict__ bv,
                bf16* __restrict__ Qb, bf16* __restrict__ Kb, bf16* __restrict__ Vb)
{
  __shared__ bf16 xs[32][CH + 16];   // [n_local][c] transposed tile, bf16
  const int b   = blockIdx.y;
  const int n0  = blockIdx.x * 32;
  const int tid = threadIdx.x;

  // Stage x[b][c][n0..n0+31] transposed into LDS (coalesced global reads).
  for (int idx = tid; idx < CH * 32; idx += 256) {
    const int c  = idx >> 5;
    const int nn = idx & 31;
    xs[nn][c] = (bf16)x[((size_t)b * CH + c) * NN + n0 + nn];
  }
  __syncthreads();

  const int wave = tid >> 5;
  const int lane = tid & 31;
  const int r  = lane & 15;
  const int hi = lane >> 4;

  // 40 o-tiles (4 q | 4 k | 32 v) x 2 n-tiles = 80 output tiles, 8 waves.
  for (int t = wave; t < 80; t += 8) {
    const int ot   = t % 40;
    const int nloc = (t / 40) * 16;
    const float* W; const float* bias; int o0; int which;
    if (ot < 4)      { W = wq; bias = bq; o0 = ot * 16;       which = 0; }
    else if (ot < 8) { W = wk; bias = bk; o0 = (ot - 4) * 16; which = 1; }
    else             { W = wv; bias = bv; o0 = (ot - 8) * 16; which = 2; }

    f32x8 acc = zero8();
    const float* wrow = W + (size_t)(o0 + r) * CH;
#pragma unroll 4
    for (int ks = 0; ks < CH / 32; ++ks) {
      const int c0 = ks * 32;
      bf16x16 a  = load_afrag_f32(wrow + c0, hi);               // W[o][c] tile
      bf16x16 bb = load_bfrag(&xs[nloc + r][c0 + hi * 16]);     // x[c][n] tile
      acc = wmma_bf16(a, bb, acc);
    }

    const int npos = n0 + nloc + r;   // D: N index = lane&15
#pragma unroll
    for (int v = 0; v < 8; ++v) {     // D: M index = v + 8*hi
      const int o = o0 + v + 8 * hi;
      const float val = acc[v] + bias[o];
      if (which == 0)      Qb[((size_t)b * NN + npos) * C8v + o] = (bf16)val;
      else if (which == 1) Kb[((size_t)b * NN + npos) * C8v + o] = (bf16)val;
      else                 Vb[((size_t)b * CH + o) * NN + npos]  = (bf16)val;
    }
  }
}

// ---------------------------------------------------------------------------
// Kernel 2: per-column softmax stats over axis i:
//   colmax[b][j] = max_i E[i,j],  colinv[b][j] = 1 / sum_i exp(E[i,j]-colmax)
// E recomputed tile-wise with WMMA; online max/sum per lane, then lane<->lane+16
// combine (wave32 shfl_xor) since D splits i across the two lane halves.
// ---------------------------------------------------------------------------
__global__ void __launch_bounds__(256)
col_softmax_stats_kernel(const bf16* __restrict__ Qb, const bf16* __restrict__ Kb,
                         float* __restrict__ colmax, float* __restrict__ colinv)
{
  const int b    = blockIdx.y;
  const int tid  = threadIdx.x;
  const int wave = tid >> 5, lane = tid & 31;
  const int r = lane & 15, hi = lane >> 4;
  const int j0 = blockIdx.x * 128 + wave * 16;   // each wave owns 16 columns

  // K B-fragments for this column block are invariant over the i loop.
  const bf16* krow = Kb + ((size_t)b * NN + j0 + r) * C8v;
  const bf16x16 kA = load_bfrag(krow + hi * 16);        // c = 0..31
  const bf16x16 kB = load_bfrag(krow + 32 + hi * 16);   // c = 32..63

  float rmax = -3.0e38f, rsum = 0.f;
  for (int it = 0; it < NN / 16; ++it) {
    const bf16* qrow = Qb + ((size_t)b * NN + it * 16 + r) * C8v;
    bf16x16 a0 = load_afrag(qrow,      hi);
    bf16x16 a1 = load_afrag(qrow + 32, hi);
    f32x8 e = zero8();
    e = wmma_bf16(a0, kA, e);
    e = wmma_bf16(a1, kB, e);

    float nm = rmax;
#pragma unroll
    for (int v = 0; v < 8; ++v) nm = fmaxf(nm, e[v]);
    float s = rsum * __expf(rmax - nm);
#pragma unroll
    for (int v = 0; v < 8; ++v) s += __expf(e[v] - nm);
    rmax = nm; rsum = s;
  }

  // Combine the two i-halves (lane l <-> l+16 hold the same column j).
  const float omax = __shfl_xor(rmax, 16, 32);
  const float osum = __shfl_xor(rsum, 16, 32);
  const float m = fmaxf(rmax, omax);
  const float s = rsum * __expf(rmax - m) + osum * __expf(omax - m);
  if (lane < 16) {
    colmax[(size_t)b * NN + j0 + lane] = m;
    colinv[(size_t)b * NN + j0 + lane] = 1.0f / s;
  }
}

// ---------------------------------------------------------------------------
// Kernel 3: fused  out[c,i] = gamma * sum_j V[c,j]*P[i,j] + x[c,i]
// One workgroup per 64-row i-block (4 i-tiles) -> each V B-fragment feeds 4
// WMMAs (4x less L2 V-traffic than a 16-row block). 512 threads = 16 waves:
//   waves 0..3 recompute the 64x64 E tile (WMMA), scale to P, stage in LDS
//   all 16 waves: 2 c-tiles x 4 i-tiles accumulators, V frags held in regs
// ---------------------------------------------------------------------------
__global__ void __launch_bounds__(512)
attn_out_kernel(const bf16* __restrict__ Qb, const bf16* __restrict__ Kb,
                const bf16* __restrict__ Vb,
                const float* __restrict__ colmax, const float* __restrict__ colinv,
                const float* __restrict__ x, const float* __restrict__ gamma,
                float* __restrict__ out)
{
  __shared__ bf16 Qtile[64][C8v + 8];  // 64 i-rows x 64 c   (9.2 KB)
  __shared__ bf16 Ptile[64][64 + 8];   // 64 i-rows x 64 j   (9.2 KB)
  const int b    = blockIdx.y;
  const int i0   = blockIdx.x * 64;
  const int tid  = threadIdx.x;
  const int wave = tid >> 5, lane = tid & 31;
  const int r = lane & 15, hi = lane >> 4;

  // Stage Q rows for this i-block once (A-fragments re-read from LDS).
  for (int idx = tid; idx < 64 * C8v; idx += 512) {
    const int i = idx >> 6, c = idx & 63;
    Qtile[i][c] = Qb[((size_t)b * NN + i0 + i) * C8v + c];
  }
  __syncthreads();

  f32x8 acc[4][2];                      // [i-tile][c-tile] = 64 VGPRs
#pragma unroll
  for (int it = 0; it < 4; ++it)
#pragma unroll
    for (int q = 0; q < 2; ++q) acc[it][q] = zero8();

  for (int jc = 0; jc < NN / 64; ++jc) {
    const int j0c = jc * 64;

    // ---- E/P phase: waves 0..3 each own a 16-column j-subtile ----
    if (wave < 4) {
      const int j0t = j0c + wave * 16;
      const bf16* krow = Kb + ((size_t)b * NN + j0t + r) * C8v;
      bf16x16 kA = load_bfrag(krow + hi * 16);        // c = 0..31
      bf16x16 kB = load_bfrag(krow + 32 + hi * 16);   // c = 32..63
      const int j = j0t + r;                          // D: column = lane&15
      const float cm = colmax[(size_t)b * NN + j];
      const float ci = colinv[(size_t)b * NN + j];
#pragma unroll
      for (int it = 0; it < 4; ++it) {
        bf16x16 aq0 = load_afrag(&Qtile[it * 16 + r][0],  hi);
        bf16x16 aq1 = load_afrag(&Qtile[it * 16 + r][32], hi);
        f32x8 e = zero8();
        e = wmma_bf16(aq0, kA, e);
        e = wmma_bf16(aq1, kB, e);
#pragma unroll
        for (int v = 0; v < 8; ++v) {                 // D: row = v + 8*hi
          const float p = __expf(e[v] - cm) * ci;
          Ptile[it * 16 + v + 8 * hi][wave * 16 + r] = (bf16)p;
        }
      }
    }
    __syncthreads();

    // ---- out phase: V B-fragments held in regs, reused by 4 i-tiles ----
    bf16x16 vA[2], vB[2];
#pragma unroll
    for (int q = 0; q < 2; ++q) {
      const int c0 = (wave * 2 + q) * 16;
      const bf16* vrow = Vb + ((size_t)b * CH + c0 + r) * NN + j0c;
      __builtin_prefetch(vrow + 64, 0, 1);            // next j-chunk of V
      vA[q] = load_bfrag(vrow + hi * 16);             // j local 0..31
      vB[q] = load_bfrag(vrow + 32 + hi * 16);        // j local 32..63
    }
#pragma unroll
    for (int it = 0; it < 4; ++it) {
      bf16x16 pA = load_afrag(&Ptile[it * 16 + r][0],  hi);
      bf16x16 pB = load_afrag(&Ptile[it * 16 + r][32], hi);
#pragma unroll
      for (int q = 0; q < 2; ++q) {
        acc[it][q] = wmma_bf16(pA, vA[q], acc[it][q]);
        acc[it][q] = wmma_bf16(pB, vB[q], acc[it][q]);
      }
    }
    __syncthreads();
  }

  // Epilogue: out = gamma*acc + x.  D: M = i offset (v+8*hi), N = c offset (r).
  const float g = gamma[0];
#pragma unroll
  for (int it = 0; it < 4; ++it) {
#pragma unroll
    for (int q = 0; q < 2; ++q) {
      const int c = (wave * 2 + q) * 16 + r;
      const size_t base = ((size_t)b * CH + c) * NN + i0 + it * 16 + 8 * hi;
#pragma unroll
      for (int v = 0; v < 8; ++v)
        out[base + v] = g * acc[it][q][v] + x[base + v];
    }
  }
}

// ---------------------------------------------------------------------------
extern "C" void kernel_launch(void* const* d_in, const int* in_sizes, int n_in,
                              void* d_out, int out_size, void* d_ws, size_t ws_size,
                              hipStream_t stream) {
  (void)in_sizes; (void)n_in; (void)out_size; (void)ws_size;
  const float* x     = (const float*)d_in[0];
  const float* wq    = (const float*)d_in[1];
  const float* bq    = (const float*)d_in[2];
  const float* wk    = (const float*)d_in[3];
  const float* bk    = (const float*)d_in[4];
  const float* wv    = (const float*)d_in[5];
  const float* bv    = (const float*)d_in[6];
  const float* gamma = (const float*)d_in[7];
  float* out = (float*)d_out;

  char* ws = (char*)d_ws;
  bf16* Qb = (bf16*)ws;        ws += (size_t)BATCH * NN * C8v * sizeof(bf16);  // 2 MB
  bf16* Kb = (bf16*)ws;        ws += (size_t)BATCH * NN * C8v * sizeof(bf16);  // 2 MB
  bf16* Vb = (bf16*)ws;        ws += (size_t)BATCH * CH * NN * sizeof(bf16);   // 16 MB
  float* colmax = (float*)ws;  ws += (size_t)BATCH * NN * sizeof(float);       // 64 KB
  float* colinv = (float*)ws;

  qkv_proj_kernel<<<dim3(NN / 32, BATCH), 256, 0, stream>>>(
      x, wq, bq, wk, bk, wv, bv, Qb, Kb, Vb);
  col_softmax_stats_kernel<<<dim3(NN / 128, BATCH), 256, 0, stream>>>(
      Qb, Kb, colmax, colinv);
  attn_out_kernel<<<dim3(NN / 64, BATCH), 512, 0, stream>>>(
      Qb, Kb, Vb, colmax, colinv, x, gamma, out);
}
